// DeepMemoryUnit_58110907515563
// MI455X (gfx1250) — compile-verified, matrix operand order checked by reading, not run on hardware
//
#include <hip/hip_runtime.h>
#include <hip/hip_bf16.h>
#include <cstdint>

// Problem dims (fixed by the reference)
#define B_  4
#define N_  2048
#define D_  1024
#define H_  16
#define HD_ 64
#define M_  (B_ * N_)   // 8192 rows for all GEMMs

typedef __attribute__((ext_vector_type(16))) _Float16 v16h;
typedef __attribute__((ext_vector_type(8)))  _Float16 v8h;
typedef __attribute__((ext_vector_type(8)))  float    v8f;

// ---------------------------------------------------------------------------
// Async global->LDS copy path (GLOBAL_LOAD_ASYNC_TO_LDS_B*, ASYNCcnt).
// Builtin signatures (from hipcc diagnostics): param0 = global (AS1) pointer
// to an int-vector of the transfer width, param1 = LDS (AS3) pointer, then
// imm offset and cpol. Guarded by __has_builtin with a sync fallback.
// ---------------------------------------------------------------------------
#define GAS __attribute__((address_space(1)))
#define LAS __attribute__((address_space(3)))

typedef __attribute__((__vector_size__(4 * sizeof(int)))) int int4v;

#if defined(__has_builtin)
#if __has_builtin(__builtin_amdgcn_global_load_async_to_lds_b128) && \
    __has_builtin(__builtin_amdgcn_global_load_async_to_lds_b32)  && \
    __has_builtin(__builtin_amdgcn_s_wait_asynccnt)
#define HAS_ASYNC_LDS 1
#endif
#endif
#ifndef HAS_ASYNC_LDS
#define HAS_ASYNC_LDS 0
#endif

__device__ __forceinline__ void cp16_g2l(void* lds, const void* g) {
#if HAS_ASYNC_LDS
  __builtin_amdgcn_global_load_async_to_lds_b128((GAS int4v*)g,
                                                 (LAS int4v*)lds, 0, 0);
#else
  *(float4*)lds = *(const float4*)g;
#endif
}
__device__ __forceinline__ void cp4_g2l(void* lds, const void* g) {
#if HAS_ASYNC_LDS
  __builtin_amdgcn_global_load_async_to_lds_b32((GAS int*)g,
                                                (LAS int*)lds, 0, 0);
#else
  *(float*)lds = *(const float*)g;
#endif
}
__device__ __forceinline__ void async_wait0() {
#if HAS_ASYNC_LDS
  __builtin_amdgcn_s_wait_asynccnt(0);
#endif
}

// ---------------------------------------------------------------------------
// Elementwise converts (fp32 -> f16), weight transpose-convert (W -> W^T f16)
// ---------------------------------------------------------------------------
__global__ void f32_to_f16_kernel(const float* __restrict__ src,
                                  _Float16* __restrict__ dst, int n) {
  int i = blockIdx.x * blockDim.x + threadIdx.x;
  if (i < n) dst[i] = (_Float16)src[i];
}

__global__ void transpose_f32_to_f16_kernel(const float* __restrict__ W,
                                            _Float16* __restrict__ Wt,
                                            int rows, int cols) {
  int i = blockIdx.x * blockDim.x + threadIdx.x;
  if (i < rows * cols) {
    int r = i / cols, c = i - r * cols;
    Wt[(size_t)c * rows + r] = (_Float16)W[i];
  }
}

// ---------------------------------------------------------------------------
// WMMA GEMM: C = epi(A @ B), A: MxK f16 row-major, Bt: NxK f16 (B transposed).
// Block tile 128x64, 8 waves, each wave: 16 rows x 64 cols (4x v_wmma 16x16x32).
// Double-buffered LDS tiles staged with async global->LDS loads: stage k+1 at
// loop top, compute WMMAs on k, then s_wait_asynccnt 0 + one barrier/iter.
// LDS pitch 40 halfs (80B): 16B-aligned rows, conflict-free fragment reads.
// ---------------------------------------------------------------------------
#define LDSP 40
enum { EPI_NONE = 0, EPI_SILU = 1, EPI_SIGLR = 2, EPI_GATE = 3 };

template <int EPI, typename OutT>
__global__ __launch_bounds__(256) void wmma_gemm_kernel(
    const _Float16* __restrict__ A,    // M x K
    const _Float16* __restrict__ Bt,   // N x K
    OutT* __restrict__ C,              // M x N
    const float* __restrict__ att,     // EPI_GATE only: M x N fp32
    int M, int Nn, int K)
{
  __shared__ _Float16 As[2][128 * LDSP];
  __shared__ _Float16 Bs[2][64 * LDSP];

  const int tid  = threadIdx.x;
  const int wave = tid >> 5;
  const int lane = tid & 31;
  const int m0 = blockIdx.x * 128;
  const int n0 = blockIdx.y * 64;

  // Per-thread staging coordinates:
  // A tile 128x32 halfs: 16 halfs/thread (two b128 chunks)
  const int ar = tid >> 1;
  const int ac = (tid & 1) << 4;
  // B tile 64x32 halfs: 8 halfs/thread (one b128 chunk)
  const int br = tid >> 2;
  const int bc = (tid & 3) << 3;
  const bool bIn = (n0 + br) < Nn;

  v8f acc[4];
#pragma unroll
  for (int i = 0; i < 4; ++i)
#pragma unroll
    for (int j = 0; j < 8; ++j) acc[i][j] = 0.f;

  auto stage = [&](int buf, int k0) {
    const _Float16* as = A + (size_t)(m0 + ar) * K + k0 + ac;
    cp16_g2l(&As[buf][ar * LDSP + ac], as);
    cp16_g2l(&As[buf][ar * LDSP + ac + 8], as + 8);
    if (bIn)
      cp16_g2l(&Bs[buf][br * LDSP + bc],
               Bt + (size_t)(n0 + br) * K + k0 + bc);
  };

  // Zero-fill out-of-bounds B rows once (both buffers); steady state is then a
  // uniform async-issue count per wave.
  if (!bIn) {
    float4 z = make_float4(0.f, 0.f, 0.f, 0.f);
    *(float4*)&Bs[0][br * LDSP + bc] = z;
    *(float4*)&Bs[1][br * LDSP + bc] = z;
  }
  stage(0, 0);
  async_wait0();
  __syncthreads();

  // Fragment addressing per CDNA5 16-bit A/B layout:
  // lane l: row = l&15, K chunks [kc, kc+8) and [kc+16, kc+24), kc = (l>>4)*8
  const int fr = lane & 15;
  const int kc = (lane >> 4) << 3;
  const int nk = K >> 5;

  for (int kt = 0; kt < nk; ++kt) {
    const int cur = kt & 1;
    if (kt + 1 < nk) stage(cur ^ 1, (kt + 1) << 5);  // async prefetch k+1

    const v8h* pa = (const v8h*)(&As[cur][(wave * 16 + fr) * LDSP + kc]);
    v8h a_lo = pa[0], a_hi = pa[2];
    v16h afrag = __builtin_shufflevector(a_lo, a_hi,
        0, 1, 2, 3, 4, 5, 6, 7, 8, 9, 10, 11, 12, 13, 14, 15);

#pragma unroll
    for (int nt = 0; nt < 4; ++nt) {
      const v8h* pb = (const v8h*)(&Bs[cur][(nt * 16 + fr) * LDSP + kc]);
      v8h b_lo = pb[0], b_hi = pb[2];
      v16h bfrag = __builtin_shufflevector(b_lo, b_hi,
          0, 1, 2, 3, 4, 5, 6, 7, 8, 9, 10, 11, 12, 13, 14, 15);
      acc[nt] = __builtin_amdgcn_wmma_f32_16x16x32_f16(
          false, afrag, false, bfrag, (short)0, acc[nt], false, false);
    }
    async_wait0();
    __syncthreads();
  }

  // Epilogue. C/D layout: VGPR r, lane l -> row = r + (l>>4)*8, col = l&15.
  const int colLane = lane & 15;
  const int rowHi   = (lane >> 4) << 3;
#pragma unroll
  for (int nt = 0; nt < 4; ++nt) {
    const int col = n0 + nt * 16 + colLane;
    if (col < Nn) {
#pragma unroll
      for (int r = 0; r < 8; ++r) {
        const int row = m0 + wave * 16 + rowHi + r;
        float x = acc[nt][r];
        float o;
        if (EPI == EPI_SILU)       o = x / (1.f + __expf(-x));
        else if (EPI == EPI_SIGLR) o = 0.01f / (1.f + __expf(-x));
        else if (EPI == EPI_GATE)  o = att[(size_t)row * Nn + col] *
                                       (1.f / (1.f + __expf(-x)));
        else                       o = x;
        C[(size_t)row * Nn + col] = (OutT)o;
      }
    }
  }
}

// ---------------------------------------------------------------------------
// Fast-weight scan: one workgroup per (b,h); W1^T, W3^T resident in LDS.
// 256 threads: (sel = W1/W3, e = output element, half = d-range half).
// Stage A: h = k^T W; Stage B: error/grad scalars; Stage C: fused rank-1
// update + o = q^T W_new; Stage D: silu-GLU + grouped RMSNorm + norm_w.
// Per-token q/k/v/lr staged into LDS via async loads; next token prefetched.
// ---------------------------------------------------------------------------
#define WPITCH 68   // 64 + 4 pad floats: 272B row stride, 16B aligned

__global__ __launch_bounds__(256) void fastweight_scan_kernel(
    const float* __restrict__ qb, const float* __restrict__ kb,
    const float* __restrict__ vb, const float* __restrict__ lrb,
    const float* __restrict__ w1_init, const float* __restrict__ w3_init,
    const float* __restrict__ norm_w, float* __restrict__ att)
{
  __shared__ __align__(16) float W[2][HD_ * WPITCH];
  __shared__ __align__(16) float kst[HD_];
  __shared__ __align__(16) float vst[HD_];
  __shared__ __align__(16) float qst[HD_];
  __shared__ float hbuf[2][HD_];
  __shared__ float gbuf[2][HD_];
  __shared__ float obuf[2][HD_];
  __shared__ float nw[HD_];
  __shared__ float red[2];
  __shared__ float lrs_s;

  const int tid = threadIdx.x;
  const int bh  = blockIdx.x;
  const int b   = bh >> 4;
  const int h   = bh & 15;

  // Load initial fast weights transposed: W[sel][e][d] = W_init[h][d][e]
  for (int i = tid; i < HD_ * HD_; i += 256) {
    int d = i >> 6, e = i & 63;
    W[0][e * WPITCH + d] = w1_init[h * HD_ * HD_ + i];
    W[1][e * WPITCH + d] = w3_init[h * HD_ * HD_ + i];
  }
  if (tid < HD_) nw[tid] = norm_w[h * HD_ + tid];
  __syncthreads();

  const int sel = tid >> 7;        // 0: W1, 1: W3
  const int r   = tid & 127;
  const int e   = r >> 1;          // output element 0..63
  const int d0  = (r & 1) << 5;    // 0 or 32

  for (int t = 0; t < N_; ++t) {
    const int base = (b * N_ + t) * D_ + h * HD_;

    // Async-stage k/v/q (3 x 64 floats as b128 chunks) + lr scalar (b32)
    if (tid < 48) {
      const int which = tid >> 4;          // 0:k 1:v 2:q
      const int e0 = (tid & 15) << 2;      // 0,4,...,60
      const float* src = (which == 0 ? kb : (which == 1 ? vb : qb)) + base + e0;
      float* dst = (which == 0 ? kst : (which == 1 ? vst : qst)) + e0;
      cp16_g2l(dst, src);
    } else if (tid == 48) {
      cp4_g2l((void*)&lrs_s, &lrb[(b * N_ + t) * H_ + h]);
    }
    if (t + 1 < N_ && tid == 224) {
      const int nb = base + D_;
      __builtin_prefetch(&kb[nb], 0, 0);  __builtin_prefetch(&kb[nb + 32], 0, 0);
      __builtin_prefetch(&vb[nb], 0, 0);  __builtin_prefetch(&vb[nb + 32], 0, 0);
      __builtin_prefetch(&qb[nb], 0, 0);  __builtin_prefetch(&qb[nb + 32], 0, 0);
    }
    async_wait0();
    __syncthreads();

    // Stage A: h1[e]/h3[e] = sum_d k[d] * W[sel][e][d]
    {
      const float4* wr = (const float4*)&W[sel][e * WPITCH + d0];
      const float4* k4 = (const float4*)&kst[d0];
      float acc = 0.f;
#pragma unroll
      for (int i = 0; i < 8; ++i) {
        float4 w4 = wr[i], kk = k4[i];
        acc = fmaf(w4.x, kk.x, acc);
        acc = fmaf(w4.y, kk.y, acc);
        acc = fmaf(w4.z, kk.z, acc);
        acc = fmaf(w4.w, kk.w, acc);
      }
      acc += __shfl_xor(acc, 1, 32);
      if ((r & 1) == 0) hbuf[sel][e] = acc;
    }
    __syncthreads();

    // Stage B: gradient scalars (pre-scaled by lr)
    if (tid < 64) {
      float h1 = hbuf[0][tid], h3 = hbuf[1][tid];
      float sg  = 1.f / (1.f + __expf(-h1));
      float s1  = h1 * sg;
      float err = s1 * h3 - vst[tid];
      float dsl = sg * (1.f + h1 * (1.f - sg));
      float l   = lrs_s;
      gbuf[0][tid] = l * (err * h3 * dsl);
      gbuf[1][tid] = l * (err * s1);
    }
    __syncthreads();

    // Stage C: W[e][d] -= c*k[d]; o[e] += q[d]*W_new[e][d]  (fused)
    {
      float4*       wr = (float4*)&W[sel][e * WPITCH + d0];
      const float4* k4 = (const float4*)&kst[d0];
      const float4* q4 = (const float4*)&qst[d0];
      const float c = gbuf[sel][e];
      float acc = 0.f;
#pragma unroll
      for (int i = 0; i < 8; ++i) {
        float4 w4 = wr[i], kk = k4[i], qq = q4[i];
        w4.x = fmaf(-c, kk.x, w4.x); acc = fmaf(qq.x, w4.x, acc);
        w4.y = fmaf(-c, kk.y, w4.y); acc = fmaf(qq.y, w4.y, acc);
        w4.z = fmaf(-c, kk.z, w4.z); acc = fmaf(qq.z, w4.z, acc);
        w4.w = fmaf(-c, kk.w, w4.w); acc = fmaf(qq.w, w4.w, acc);
        wr[i] = w4;
      }
      acc += __shfl_xor(acc, 1, 32);
      if ((r & 1) == 0) obuf[sel][e] = acc;
    }
    __syncthreads();

    // Stage D: silu(o1)*o3, grouped RMSNorm over HD, norm_w scale
    float val = 0.f;
    if (tid < 64) {
      float o1 = obuf[0][tid], o3 = obuf[1][tid];
      val = (o1 / (1.f + __expf(-o1))) * o3;
      float sq = val * val;
#pragma unroll
      for (int off = 16; off > 0; off >>= 1) sq += __shfl_xor(sq, off, 32);
      if ((tid & 31) == 0) red[tid >> 5] = sq;
    }
    __syncthreads();
    if (tid < 64) {
      float rms = rsqrtf((red[0] + red[1]) * (1.f / 64.f) + 1e-6f);
      att[base + tid] = val * rms * nw[tid];
    }
    __syncthreads();
  }
}

// ---------------------------------------------------------------------------
// Launch: converts -> projection GEMMs -> scan -> gate GEMM -> output GEMM
// ---------------------------------------------------------------------------
extern "C" void kernel_launch(void* const* d_in, const int* in_sizes, int n_in,
                              void* d_out, int out_size, void* d_ws, size_t ws_size,
                              hipStream_t stream) {
  (void)in_sizes; (void)n_in; (void)out_size; (void)ws_size;

  const float* x   = (const float*)d_in[0];
  const float* Wq  = (const float*)d_in[1];
  const float* Wk  = (const float*)d_in[2];
  const float* Wv  = (const float*)d_in[3];
  const float* Wo  = (const float*)d_in[4];
  const float* w1i = (const float*)d_in[5];
  const float* w3i = (const float*)d_in[6];
  const float* Wlr = (const float*)d_in[7];
  const float* nw  = (const float*)d_in[8];
  const float* Wg1 = (const float*)d_in[9];
  const float* Wg2 = (const float*)d_in[10];

  char* ws = (char*)d_ws;
  size_t off = 0;
  auto take = [&](size_t bytes) -> void* {
    void* p = ws + off;
    off = (off + bytes + 255) & ~(size_t)255;
    return p;
  };

  _Float16* xh   = (_Float16*)take((size_t)M_ * D_ * 2);
  _Float16* Wqt  = (_Float16*)take((size_t)D_ * D_ * 2);
  _Float16* Wkt  = (_Float16*)take((size_t)D_ * D_ * 2);
  _Float16* Wvt  = (_Float16*)take((size_t)D_ * D_ * 2);
  _Float16* Wot  = (_Float16*)take((size_t)D_ * D_ * 2);
  _Float16* Wlrt = (_Float16*)take((size_t)D_ * H_ * 2);
  _Float16* Wg1t = (_Float16*)take((size_t)D_ * HD_ * 2);
  _Float16* Wg2t = (_Float16*)take((size_t)HD_ * D_ * 2);
  float* qbuf  = (float*)take((size_t)M_ * D_ * 4);
  float* kbuf  = (float*)take((size_t)M_ * D_ * 4);
  float* vbuf  = (float*)take((size_t)M_ * D_ * 4);
  float* lrbuf = (float*)take((size_t)M_ * H_ * 4);
  _Float16* gbh  = (_Float16*)take((size_t)M_ * HD_ * 2);
  float* attb  = (float*)take((size_t)M_ * D_ * 4);
  _Float16* yh = xh;   // reuse: xh is dead after projection GEMMs

  const int cvt = 256;
  f32_to_f16_kernel<<<(M_ * D_ + cvt - 1) / cvt, cvt, 0, stream>>>(x, xh, M_ * D_);
  transpose_f32_to_f16_kernel<<<(D_ * D_ + cvt - 1) / cvt, cvt, 0, stream>>>(Wq, Wqt, D_, D_);
  transpose_f32_to_f16_kernel<<<(D_ * D_ + cvt - 1) / cvt, cvt, 0, stream>>>(Wk, Wkt, D_, D_);
  transpose_f32_to_f16_kernel<<<(D_ * D_ + cvt - 1) / cvt, cvt, 0, stream>>>(Wv, Wvt, D_, D_);
  transpose_f32_to_f16_kernel<<<(D_ * D_ + cvt - 1) / cvt, cvt, 0, stream>>>(Wo, Wot, D_, D_);
  transpose_f32_to_f16_kernel<<<(D_ * H_ + cvt - 1) / cvt, cvt, 0, stream>>>(Wlr, Wlrt, D_, H_);
  transpose_f32_to_f16_kernel<<<(D_ * HD_ + cvt - 1) / cvt, cvt, 0, stream>>>(Wg1, Wg1t, D_, HD_);
  transpose_f32_to_f16_kernel<<<(HD_ * D_ + cvt - 1) / cvt, cvt, 0, stream>>>(Wg2, Wg2t, HD_, D_);

  dim3 gBig(M_ / 128, D_ / 64);     // 64 x 16 blocks
  dim3 gNar(M_ / 128, 1);           // narrow-N GEMMs

  wmma_gemm_kernel<EPI_SILU, float><<<gBig, 256, 0, stream>>>(xh, Wqt, qbuf, nullptr, M_, D_, D_);
  wmma_gemm_kernel<EPI_SILU, float><<<gBig, 256, 0, stream>>>(xh, Wkt, kbuf, nullptr, M_, D_, D_);
  wmma_gemm_kernel<EPI_NONE, float><<<gBig, 256, 0, stream>>>(xh, Wvt, vbuf, nullptr, M_, D_, D_);
  wmma_gemm_kernel<EPI_SIGLR, float><<<gNar, 256, 0, stream>>>(xh, Wlrt, lrbuf, nullptr, M_, H_, D_);
  wmma_gemm_kernel<EPI_NONE, _Float16><<<gNar, 256, 0, stream>>>(xh, Wg1t, gbh, nullptr, M_, HD_, D_);

  fastweight_scan_kernel<<<B_ * H_, 256, 0, stream>>>(qbuf, kbuf, vbuf, lrbuf,
                                                      w1i, w3i, nw, attb);

  // gate = sigmoid(gbott @ Wg2); y = att * gate  (f16 out, overwrites xh)
  wmma_gemm_kernel<EPI_GATE, _Float16><<<gBig, 256, 0, stream>>>(gbh, Wg2t, yh, attb, M_, D_, HD_);
  // final: out = y @ Wo (fp32)
  wmma_gemm_kernel<EPI_NONE, float><<<gBig, 256, 0, stream>>>(yh, Wot, (float*)d_out, nullptr, M_, D_, D_);
}
